// KVBlockCompressor_10840497455462
// MI455X (gfx1250) — matrix operand order repaired and checked by reading, not compiled, not run on hardware
//
#include <hip/hip_runtime.h>

#define NB 1024
#define NH 8
#define BS 64
#define HD 128
#define TWO_D 256
#define HID 64

typedef __bf16 bf16_t;
typedef __attribute__((ext_vector_type(16))) __bf16 v16bf;
typedef __attribute__((ext_vector_type(8)))  __bf16 v8bf;
typedef __attribute__((ext_vector_type(4)))  __bf16 v4bf;
typedef __attribute__((ext_vector_type(8)))  float  v8f;

// LDS row strides (elements), padded for bank-conflict-free access while
// keeping 16-byte alignment for b128 LDS ops (multiples of 8 elements).
#define XSTR 264   // X tile [64 rows][256 cols] bf16
#define WSTR 264   // (fallback only) W1^T [64][256] bf16
#define HSTR 68    // h [64 rows][64 cols] f32

// ---------------------------------------------------------------------------
// Prep kernel: W1 [256][64] f32 -> W1^T [64][256] bf16 in workspace (32 KB).
// Run once per launch; dispatch-invariant data, L2/WGP$-resident afterwards.
// ---------------------------------------------------------------------------
__global__ __launch_bounds__(256)
void w1_transpose_bf16_kernel(const float* __restrict__ W1,
                              bf16_t* __restrict__ w1t)
{
    int idx = blockIdx.x * 256 + threadIdx.x;   // 0..16383, coalesced read
    int k = idx >> 6;
    int n = idx & 63;
    w1t[n * TWO_D + k] = (__bf16)W1[idx];
}

// ---------------------------------------------------------------------------
// Primary kernel: B fragments straight from global bf16 W1^T (no LDS for W1)
// LDS = 33,792 (X) + 17,408 (h) + ~2 KB small = ~52 KB -> 6 WGs per 320KB WGP
// ---------------------------------------------------------------------------
__global__ __launch_bounds__(256)
void kv_compress_ws_kernel(const float* __restrict__ key_raw,
                           const float* __restrict__ value,
                           const bf16_t* __restrict__ w1t,
                           const float* __restrict__ W2,
                           float* __restrict__ out)
{
    __shared__ __align__(16) bf16_t lds_x[BS * XSTR];   // 33,792 B
    __shared__ __align__(16) float  lds_h[BS * HSTR];   // 17,408 B
    __shared__ __align__(16) float  lds_part[BS * 4];   // score partials
    __shared__ float lds_exp[BS];
    __shared__ float lds_wgt[BS];
    __shared__ float lds_w2[HID];

    const int tid = threadIdx.x;
    const int bh  = blockIdx.x;
    const size_t tile_base = (size_t)bh * BS * HD;

    // ---- Stage X = [K|V] -> LDS bf16, coalesced float4 sweeps ----
    {
        const float4* kp = (const float4*)(key_raw + tile_base);
        const float4* vp = (const float4*)(value   + tile_base);
        #pragma unroll
        for (int i = 0; i < 8; ++i) {
            int e4 = tid + i * 256;
            int e  = e4 << 2;
            int s  = e >> 7;
            int dd = e & 127;
            float4 kf = kp[e4];
            float4 vf = vp[e4];
            v4bf kb = { (__bf16)kf.x, (__bf16)kf.y, (__bf16)kf.z, (__bf16)kf.w };
            v4bf vb = { (__bf16)vf.x, (__bf16)vf.y, (__bf16)vf.z, (__bf16)vf.w };
            *(v4bf*)&lds_x[s * XSTR + dd]       = kb;
            *(v4bf*)&lds_x[s * XSTR + 128 + dd] = vb;
        }
    }
    if (tid < HID) lds_w2[tid] = W2[tid];
    __syncthreads();

    // ---- GEMM1: h = silu(X @ W1) via bf16 WMMA, A from LDS, B from global ----
    const int lane = tid & 31;
    const int wv   = tid >> 5;
    const int mi   = wv & 3;
    const int nb0  = (wv >> 2) * 2;
    const int rsel = lane >> 4;
    const int rloc = lane & 15;
    const int arow = mi * 16 + rloc;
    const int aoff = rsel * 8;
    const int boff = rsel * 8;
    const bf16_t* bp0 = w1t + (size_t)((nb0 + 0) * 16 + rloc) * TWO_D + boff;
    const bf16_t* bp1 = w1t + (size_t)((nb0 + 1) * 16 + rloc) * TWO_D + boff;

    v8f acc0 = {};
    v8f acc1 = {};
    #pragma unroll
    for (int kk = 0; kk < TWO_D; kk += 32) {
        v8bf alo = *(const v8bf*)&lds_x[arow * XSTR + kk + aoff];
        v8bf ahi = *(const v8bf*)&lds_x[arow * XSTR + kk + 16 + aoff];
        v16bf a = __builtin_shufflevector(alo, ahi,
            0,1,2,3,4,5,6,7,8,9,10,11,12,13,14,15);

        v8bf b0lo = *(const v8bf*)(bp0 + kk);
        v8bf b0hi = *(const v8bf*)(bp0 + kk + 16);
        v16bf b0 = __builtin_shufflevector(b0lo, b0hi,
            0,1,2,3,4,5,6,7,8,9,10,11,12,13,14,15);

        v8bf b1lo = *(const v8bf*)(bp1 + kk);
        v8bf b1hi = *(const v8bf*)(bp1 + kk + 16);
        v16bf b1 = __builtin_shufflevector(b1lo, b1hi,
            0,1,2,3,4,5,6,7,8,9,10,11,12,13,14,15);

        acc0 = __builtin_amdgcn_wmma_f32_16x16x32_bf16(
            false, a, false, b0, (short)0, acc0, false, false);
        acc1 = __builtin_amdgcn_wmma_f32_16x16x32_bf16(
            false, a, false, b1, (short)0, acc1, false, false);
    }

    #pragma unroll
    for (int i = 0; i < 8; ++i) {
        int row = mi * 16 + i + rsel * 8;
        float x0 = acc0[i];
        float x1 = acc1[i];
        x0 = x0 / (1.0f + __expf(-x0));
        x1 = x1 / (1.0f + __expf(-x1));
        lds_h[row * HSTR + (nb0 + 0) * 16 + rloc] = x0;
        lds_h[row * HSTR + (nb0 + 1) * 16 + rloc] = x1;
    }
    __syncthreads();

    // ---- scores = h @ W2, parallel over all 256 threads (4 partials/row) ----
    {
        int r = tid >> 2;
        int q = tid & 3;
        float p = 0.0f;
        #pragma unroll
        for (int c = 0; c < 16; ++c)
            p += lds_h[r * HSTR + q * 16 + c] * lds_w2[q * 16 + c];
        lds_part[tid] = p;
    }
    __syncthreads();
    if (tid < BS) {
        float4 pp = *(const float4*)&lds_part[tid * 4];
        lds_wgt[tid] = pp.x + pp.y + pp.z + pp.w;   // raw score
    }
    __syncthreads();
    if (tid < BS) {
        float m = -3.0e38f;
        #pragma unroll 8
        for (int c = 0; c < BS; ++c) m = fmaxf(m, lds_wgt[c]);
        lds_exp[tid] = __expf(lds_wgt[tid] - m);
    }
    __syncthreads();
    if (tid < BS) {
        float sum = 0.0f;
        #pragma unroll 8
        for (int c = 0; c < BS; ++c) sum += lds_exp[c];
        lds_wgt[tid] = lds_exp[tid] / sum;          // final weights
    }
    __syncthreads();

    // ---- weighted summaries: thread t reduces column t of X ----
    {
        float acc = 0.0f;
        #pragma unroll 8
        for (int s = 0; s < BS; ++s)
            acc += lds_wgt[s] * (float)lds_x[s * XSTR + tid];
        const int half = tid >> 7;
        const int dd   = tid & 127;
        size_t o = (size_t)bh * HD + dd;
        if (half) o += (size_t)NB * NH * HD;
        out[o] = acc;
    }
}

// ---------------------------------------------------------------------------
// Fallback (self-contained, used only if ws_size < 32 KB): Round-1 kernel.
// ---------------------------------------------------------------------------
__global__ __launch_bounds__(256)
void kv_compress_fallback_kernel(const float* __restrict__ key_raw,
                                 const float* __restrict__ value,
                                 const float* __restrict__ W1,
                                 const float* __restrict__ W2,
                                 float* __restrict__ out)
{
    __shared__ __align__(16) bf16_t lds_x[BS * XSTR];
    __shared__ __align__(16) bf16_t lds_w1t[HID * WSTR];
    __shared__ __align__(16) float  lds_h[BS * HSTR];
    __shared__ float lds_exp[BS];
    __shared__ float lds_wgt[BS];
    __shared__ float lds_w2[HID];

    const int tid = threadIdx.x;
    const int bh  = blockIdx.x;
    const size_t tile_base = (size_t)bh * BS * HD;

    {
        const float4* kp = (const float4*)(key_raw + tile_base);
        const float4* vp = (const float4*)(value   + tile_base);
        #pragma unroll
        for (int i = 0; i < 8; ++i) {
            int e4 = tid + i * 256;
            int e  = e4 << 2;
            int s  = e >> 7;
            int dd = e & 127;
            float4 kf = kp[e4];
            float4 vf = vp[e4];
            v4bf kb = { (__bf16)kf.x, (__bf16)kf.y, (__bf16)kf.z, (__bf16)kf.w };
            v4bf vb = { (__bf16)vf.x, (__bf16)vf.y, (__bf16)vf.z, (__bf16)vf.w };
            *(v4bf*)&lds_x[s * XSTR + dd]       = kb;
            *(v4bf*)&lds_x[s * XSTR + 128 + dd] = vb;
        }
        #pragma unroll 4
        for (int i = 0; i < 64; ++i) {
            int idx = tid + i * 256;
            int k = idx >> 6;
            int n = idx & 63;
            lds_w1t[n * WSTR + k] = (__bf16)W1[idx];
        }
        if (tid < HID) lds_w2[tid] = W2[tid];
    }
    __syncthreads();

    const int lane = tid & 31;
    const int wv   = tid >> 5;
    const int mi   = wv & 3;
    const int nb0  = (wv >> 2) * 2;
    const int rsel = lane >> 4;
    const int rloc = lane & 15;
    const int arow = mi * 16 + rloc;
    const int aoff = rsel * 8;
    const int bn0  = (nb0 + 0) * 16 + rloc;
    const int bn1  = (nb0 + 1) * 16 + rloc;
    const int boff = rsel * 8;

    v8f acc0 = {};
    v8f acc1 = {};
    #pragma unroll
    for (int kk = 0; kk < TWO_D; kk += 32) {
        v8bf alo = *(const v8bf*)&lds_x[arow * XSTR + kk + aoff];
        v8bf ahi = *(const v8bf*)&lds_x[arow * XSTR + kk + 16 + aoff];
        v16bf a = __builtin_shufflevector(alo, ahi,
            0,1,2,3,4,5,6,7,8,9,10,11,12,13,14,15);
        v8bf b0lo = *(const v8bf*)&lds_w1t[bn0 * WSTR + kk + boff];
        v8bf b0hi = *(const v8bf*)&lds_w1t[bn0 * WSTR + kk + boff + 16];
        v16bf b0 = __builtin_shufflevector(b0lo, b0hi,
            0,1,2,3,4,5,6,7,8,9,10,11,12,13,14,15);
        v8bf b1lo = *(const v8bf*)&lds_w1t[bn1 * WSTR + kk + boff];
        v8bf b1hi = *(const v8bf*)&lds_w1t[bn1 * WSTR + kk + boff + 16];
        v16bf b1 = __builtin_shufflevector(b1lo, b1hi,
            0,1,2,3,4,5,6,7,8,9,10,11,12,13,14,15);
        acc0 = __builtin_amdgcn_wmma_f32_16x16x32_bf16(
            false, a, false, b0, (short)0, acc0, false, false);
        acc1 = __builtin_amdgcn_wmma_f32_16x16x32_bf16(
            false, a, false, b1, (short)0, acc1, false, false);
    }

    #pragma unroll
    for (int i = 0; i < 8; ++i) {
        int row = mi * 16 + i + rsel * 8;
        float x0 = acc0[i];
        float x1 = acc1[i];
        x0 = x0 / (1.0f + __expf(-x0));
        x1 = x1 / (1.0f + __expf(-x1));
        lds_h[row * HSTR + (nb0 + 0) * 16 + rloc] = x0;
        lds_h[row * HSTR + (nb0 + 1) * 16 + rloc] = x1;
    }
    __syncthreads();

    if (tid < BS) {
        float s = 0.0f;
        #pragma unroll 8
        for (int c = 0; c < HID; ++c)
            s += lds_h[tid * HSTR + c] * lds_w2[c];
        lds_wgt[tid] = s;
    }
    __syncthreads();
    if (tid < BS) {
        float m = -3.0e38f;
        #pragma unroll 8
        for (int c = 0; c < BS; ++c) m = fmaxf(m, lds_wgt[c]);
        lds_exp[tid] = __expf(lds_wgt[tid] - m);
    }
    __syncthreads();
    if (tid < BS) {
        float sum = 0.0f;
        #pragma unroll 8
        for (int c = 0; c < BS; ++c) sum += lds_exp[c];
        lds_wgt[tid] = lds_exp[tid] / sum;
    }
    __syncthreads();

    {
        float acc = 0.0f;
        #pragma unroll 8
        for (int s = 0; s < BS; ++s)
            acc += lds_wgt[s] * (float)lds_x[s * XSTR + tid];
        const int half = tid >> 7;
        const int dd   = tid & 127;
        size_t o = (size_t)bh * HD + dd;
        if (half) o += (size_t)NB * NH * HD;
        out[o] = acc;
    }
}

extern "C" void kernel_launch(void* const* d_in, const int* in_sizes, int n_in,
                              void* d_out, int out_size, void* d_ws, size_t ws_size,
                              hipStream_t stream) {
    const float* key_raw = (const float*)d_in[0];
    const float* value   = (const float*)d_in[1];
    const float* W1      = (const float*)d_in[2];
    const float* W2      = (const float*)d_in[3];
    float* out = (float*)d_out;

    const size_t w1t_bytes = (size_t)HID * TWO_D * sizeof(bf16_t);  // 32 KB
    dim3 grid(NB * NH);
    dim3 block(256);

    if (d_ws != nullptr && ws_size >= w1t_bytes) {
        bf16_t* w1t = (bf16_t*)d_ws;
        w1_transpose_bf16_kernel<<<dim3(64), block, 0, stream>>>(W1, w1t);
        kv_compress_ws_kernel<<<grid, block, 0, stream>>>(key_raw, value, w1t, W2, out);
    } else {
        kv_compress_fallback_kernel<<<grid, block, 0, stream>>>(key_raw, value, W1, W2, out);
    }
}
